// Mamba3D_532575945299
// MI455X (gfx1250) — matrix operand order repaired and checked by reading, not compiled, not run on hardware
//
#include <hip/hip_runtime.h>
#include <hip/hip_bf16.h>
#include <math.h>

// ---------------- problem constants ----------------
#define Bn   8
#define Gn   1024
#define Cn   384
#define D2n  768
#define DINn 768
#define DSTATEn 16
#define DTRANKn 24
#define DCONVn 4
#define Kn   8
#define MROWS (Bn * Gn)            // 8192
#define XDW  (DTRANKn + 2*DSTATEn) // 56 logical
#define XDP  64                    // padded ld for xd

typedef __attribute__((ext_vector_type(16))) _Float16 v16h;
typedef __attribute__((ext_vector_type(8)))  _Float16 v8h;
typedef __attribute__((ext_vector_type(8)))  float    v8f;

__device__ __forceinline__ float silu_f(float x) { return x / (1.f + expf(-x)); }

// ---------------- f32 -> f16 bulk convert (n % 4 == 0) ----------------
__global__ void cvt_f16_kernel(const float* __restrict__ in, _Float16* __restrict__ out, long n) {
    long t = ((long)blockIdx.x * blockDim.x + threadIdx.x) * 4;
    if (t >= n) return;
    float4 v = *(const float4*)&in[t];
    _Float16* o = &out[t];
    o[0] = (_Float16)v.x; o[1] = (_Float16)v.y;
    o[2] = (_Float16)v.z; o[3] = (_Float16)v.w;
}

// ---------------- 1) adaLN modulation: mods = silu(temd) @ adaln_w + adaln_b ----------------
__global__ void mod_kernel(const float* __restrict__ temd, const float* __restrict__ aw,
                           const float* __restrict__ ab, float* __restrict__ mods) {
    int j = blockIdx.x * blockDim.x + threadIdx.x;   // over B * 6C
    if (j >= Bn * 6 * Cn) return;
    int b = j / (6 * Cn), col = j % (6 * Cn);
    float acc = ab[col];
    for (int c = 0; c < Cn; ++c) {
        float t = temd[b * Cn + c];
        acc += silu_f(t) * aw[c * (6 * Cn) + col];
    }
    mods[j] = acc;
}

// ---------------- 2) LN + modulate (wave32 per row) ----------------
__global__ void lnmod_kernel(const float* __restrict__ in, const float* __restrict__ g,
                             const float* __restrict__ bvec, const float* __restrict__ mods,
                             int shChunk, int scChunk, float* __restrict__ out) {
    int wave = threadIdx.x >> 5, lane = threadIdx.x & 31;
    long row = (long)blockIdx.x * 8 + wave;          // B*G rows
    int b = (int)(row >> 10);
    const float* rp = in + row * Cn;
    float v[12]; float s = 0.f, ss = 0.f;
#pragma unroll
    for (int t = 0; t < 12; ++t) { float x = rp[lane + t * 32]; v[t] = x; s += x; ss += x * x; }
#pragma unroll
    for (int off = 16; off > 0; off >>= 1) { s += __shfl_xor(s, off, 32); ss += __shfl_xor(ss, off, 32); }
    float mean = s * (1.f / Cn);
    float var  = ss * (1.f / Cn) - mean * mean;
    float r = rsqrtf(var + 1e-5f);
#pragma unroll
    for (int t = 0; t < 12; ++t) {
        int c = lane + t * 32;
        float h = (v[t] - mean) * r * g[c] + bvec[c];
        out[row * Cn + c] = h * (1.f + mods[b * 6 * Cn + scChunk * Cn + c])
                          + mods[b * 6 * Cn + shChunk * Cn + c];
    }
}

// ---------------- 3) KNN top-8 (smallest d2, stable ties) ----------------
__global__ void knn_kernel(const float* __restrict__ center, int* __restrict__ idx_out) {
    __shared__ float cs[Gn * 3];
    __shared__ float sq[Gn];
    int b = blockIdx.y;
    const float* cb = center + (long)b * Gn * 3;
    for (int t = threadIdx.x; t < Gn * 3; t += blockDim.x) cs[t] = cb[t];
    __syncthreads();
    for (int t = threadIdx.x; t < Gn; t += blockDim.x)
        sq[t] = cs[3*t]*cs[3*t] + cs[3*t+1]*cs[3*t+1] + cs[3*t+2]*cs[3*t+2];
    __syncthreads();
    int i = blockIdx.x * blockDim.x + threadIdx.x;
    if (i >= Gn) return;
    float xi = cs[3*i], yi = cs[3*i+1], zi = cs[3*i+2], sqi = sq[i];
    float bestd[Kn]; int besti[Kn];
#pragma unroll
    for (int k = 0; k < Kn; ++k) { bestd[k] = 3.0e38f; besti[k] = 0; }
    for (int j = 0; j < Gn; ++j) {
        float d2 = sqi + sq[j] - 2.f * (xi*cs[3*j] + yi*cs[3*j+1] + zi*cs[3*j+2]);
        if (d2 < bestd[Kn-1]) {
            int p = Kn - 1;
            while (p > 0 && d2 < bestd[p-1]) { bestd[p] = bestd[p-1]; besti[p] = besti[p-1]; --p; }
            bestd[p] = d2; besti[p] = j;
        }
    }
#pragma unroll
    for (int k = 0; k < Kn; ++k) idx_out[((long)b * Gn + i) * Kn + k] = besti[k];
}

// ---------------- 4) global std of diff (two-pass deterministic reduction) ----------------
#define STAT_BLOCKS 1024
__global__ void diffstat_kernel(const float* __restrict__ h1, const int* __restrict__ idx,
                                float* __restrict__ partial) {
    const long total = (long)Bn * Gn * Kn * Cn;
    long start = (long)blockIdx.x * blockDim.x + threadIdx.x;
    long stride = (long)gridDim.x * blockDim.x;
    float s = 0.f, ss = 0.f;
    for (long e = start; e < total; e += stride) {
        int c = (int)(e % Cn);
        long r = e / Cn;
        int k = (int)(r % Kn);
        long row = r / Kn;                       // b*Gn+g
        long b = row >> 10;
        int nb = idx[row * Kn + k];
        float dv = h1[(b * Gn + nb) * Cn + c] - h1[row * Cn + c];
        s += dv; ss += dv * dv;
    }
    __shared__ float rs[256], rss[256];
    rs[threadIdx.x] = s; rss[threadIdx.x] = ss;
    __syncthreads();
    for (int o = 128; o > 0; o >>= 1) {
        if ((int)threadIdx.x < o) { rs[threadIdx.x] += rs[threadIdx.x+o]; rss[threadIdx.x] += rss[threadIdx.x+o]; }
        __syncthreads();
    }
    if (threadIdx.x == 0) { partial[blockIdx.x*2] = rs[0]; partial[blockIdx.x*2+1] = rss[0]; }
}

__global__ void diffstat_final(const float* __restrict__ partial, float* __restrict__ stats) {
    __shared__ float rs[256], rss[256];
    float s = 0.f, ss = 0.f;
    for (int i = threadIdx.x; i < STAT_BLOCKS; i += 256) { s += partial[2*i]; ss += partial[2*i+1]; }
    rs[threadIdx.x] = s; rss[threadIdx.x] = ss;
    __syncthreads();
    for (int o = 128; o > 0; o >>= 1) {
        if ((int)threadIdx.x < o) { rs[threadIdx.x] += rs[threadIdx.x+o]; rss[threadIdx.x] += rss[threadIdx.x+o]; }
        __syncthreads();
    }
    if (threadIdx.x == 0) {
        const float N = (float)((long)Bn * Gn * Kn * Cn);
        float var = (rss[0] - rs[0] * rs[0] / N) / (N - 1.f);
        stats[0] = 1.f / (sqrtf(var) + 1e-5f);
    }
}

// ---------------- 5) local softmax-pool + prenorm LN -> lcn (B,G,768) ----------------
__global__ void lc_kernel(const float* __restrict__ h1, const int* __restrict__ idx,
                          const float* __restrict__ stats,
                          const float* __restrict__ alpha, const float* __restrict__ beta,
                          const float* __restrict__ png, const float* __restrict__ pnb,
                          float* __restrict__ lcn) {
    long row = blockIdx.x;                  // b*Gn+g
    long b = row >> 10;
    __shared__ int nb[Kn];
    __shared__ float rs[256], rss[256];
    if (threadIdx.x < Kn) nb[threadIdx.x] = idx[row * Kn + threadIdx.x];
    __syncthreads();
    float inv = stats[0];
    float vals[3]; float s = 0.f, ss = 0.f;
#pragma unroll
    for (int t = 0; t < 3; ++t) {
        int cc = threadIdx.x + t * 256;     // 0..767
        float v;
        if (cc < Cn) {
            float hc = h1[row * Cn + cc];
            float al = alpha[cc], be = beta[cc];
            float se = 0.f, swe = 0.f;
#pragma unroll
            for (int k = 0; k < Kn; ++k) {
                float nv = h1[(b * Gn + nb[k]) * Cn + cc];
                float cat = al * ((nv - hc) * inv) + be;
                float e = expf(cat);
                se += e; swe += cat * e;
            }
            v = swe / se;
        } else {
            int c = cc - Cn;
            v = alpha[cc] * h1[row * Cn + c] + beta[cc];  // identical across K -> pool is identity
        }
        vals[t] = v; s += v; ss += v * v;
    }
    rs[threadIdx.x] = s; rss[threadIdx.x] = ss;
    __syncthreads();
    for (int o = 128; o > 0; o >>= 1) {
        if ((int)threadIdx.x < o) { rs[threadIdx.x] += rs[threadIdx.x+o]; rss[threadIdx.x] += rss[threadIdx.x+o]; }
        __syncthreads();
    }
    float mean = rs[0] * (1.f / D2n);
    float var  = rss[0] * (1.f / D2n) - mean * mean;
    float r = rsqrtf(var + 1e-5f);
#pragma unroll
    for (int t = 0; t < 3; ++t) {
        int cc = threadIdx.x + t * 256;
        lcn[row * D2n + cc] = (vals[t] - mean) * r * png[cc] + pnb[cc];
    }
}

// ---------------- weight padding -> f16 (makes GEMM dims regular) ----------------
__global__ void pad_xw_kernel(const float* __restrict__ in, _Float16* __restrict__ out) {
    int t = blockIdx.x * blockDim.x + threadIdx.x;   // 768*64
    if (t >= DINn * XDP) return;
    int k = t >> 6, n = t & 63;
    out[t] = (n < XDW) ? (_Float16)in[k * XDW + n] : (_Float16)0.f;
}
__global__ void pad_dw_kernel(const float* __restrict__ in, _Float16* __restrict__ out) {
    int t = blockIdx.x * blockDim.x + threadIdx.x;   // 64*768
    if (t >= 64 * DINn) return;
    int k = t / DINn, n = t % DINn;
    out[t] = (k < DTRANKn) ? (_Float16)in[k * DINn + n] : (_Float16)0.f;
}

// ---------------- LDS-tiled WMMA GEMM (f16 inputs): C[M,N] = A[M,K](lda) * W[K,N] ----------------
// Requirements: M % 128 == 0, N % 16 == 0, K % 64 == 0, lda % 8 == 0. No bounds checks.
// A tile (128x64 f16) staged via CDNA5 async global->LDS copies (ASYNCcnt);
// B tile (64x16 f16) staged transposed via DS stores.
#define BM 128
#define KS 64
#define LSTR 72   // LDS row stride in halves (16B aligned, breaks bank conflicts)
__global__ void __launch_bounds__(256) gemm_wmma_kernel(const _Float16* __restrict__ A, int lda,
                                                        const _Float16* __restrict__ W,
                                                        float* __restrict__ C, int ldc,
                                                        int N, int K) {
    __shared__ __align__(16) _Float16 lA[BM * LSTR];
    __shared__ __align__(16) _Float16 lB[16 * LSTR];
    const int lane = threadIdx.x & 31, wave = threadIdx.x >> 5;
    const int nBase = blockIdx.x * 16;
    const long mBase = (long)blockIdx.y * BM;
    const int fm = lane & 15;
    const int kOffA = (lane >> 4) * 8;   // A frag: K halves by lane>>4 within 32-wide subtile
    const int kOffB = (lane >> 4) * 8;   // B frag: K 8-groups by lane>>4 within 32-wide subtile (same as A)
    v8f acc = {};

    // two 16-byte async chunks per thread per K-step: 512 chunks cover 64x64 halves (rows 0..63)
    const int q0 = threadIdx.x, q1 = threadIdx.x + 256;
    const int r0 = q0 >> 3, c0 = (q0 & 7) * 8;
    const int r1 = q1 >> 3, c1 = (q1 & 7) * 8;
    unsigned dst0 = (unsigned)(uintptr_t)&lA[r0 * LSTR + c0];  // low 32 bits of flat LDS addr = LDS offset
    unsigned dst1 = (unsigned)(uintptr_t)&lA[r1 * LSTR + c1];

    for (int k0 = 0; k0 < K; k0 += KS) {
        __syncthreads();                       // previous tile fully consumed
        const _Float16* s0 = &A[(mBase + r0) * lda + k0 + c0];
        const _Float16* s1 = &A[(mBase + r1) * lda + k0 + c1];
        asm volatile("global_load_async_to_lds_b128 %0, %1, off" :: "v"(dst0), "v"(s0) : "memory");
        asm volatile("global_load_async_to_lds_b128 %0, %1, off" :: "v"(dst1), "v"(s1) : "memory");
        // stage B: 64 x 16 f16, stored transposed [n][k]
#pragma unroll
        for (int p = 0; p < 4; ++p) {
            int idx = threadIdx.x + p * 256;
            int kk = idx >> 4, nn = idx & 15;
            lB[nn * LSTR + kk] = W[(long)(k0 + kk) * N + nBase + nn];
        }
        asm volatile("s_wait_asynccnt 0x0" ::: "memory");   // own async copies landed
        __syncthreads();                                    // all waves' copies visible
#pragma unroll
        for (int h = 0; h < 2; ++h) {
            const _Float16* pa = &lA[(wave * 16 + fm) * LSTR + h * 32 + kOffA];
            v8h a0 = *(const v8h*)pa;
            v8h a1 = *(const v8h*)(pa + 16);
            v16h a;
#pragma unroll
            for (int e = 0; e < 8; ++e) { a[e] = a0[e]; a[e + 8] = a1[e]; }
            const _Float16* pb = &lB[fm * LSTR + h * 32 + kOffB];
            v8h b0 = *(const v8h*)pb;
            v8h b1 = *(const v8h*)(pb + 16);
            v16h bf;
#pragma unroll
            for (int e = 0; e < 8; ++e) { bf[e] = b0[e]; bf[e + 8] = b1[e]; }
            acc = __builtin_amdgcn_wmma_f32_16x16x32_f16(false, a, false, bf,
                                                         (short)0, acc, false, false);
        }
    }
    const long mOut = mBase + wave * 16 + ((lane >> 4) << 3);
    const int nCol = nBase + fm;
#pragma unroll
    for (int i = 0; i < 8; ++i) C[(mOut + i) * ldc + nCol] = acc[i];
}

// ---------------- epilogue 1: x1 = x + g_msa * silu(raw + mlp_b) ----------------
__global__ void epi1_kernel(const float* __restrict__ x, const float* __restrict__ raw,
                            const float* __restrict__ mlp_b, const float* __restrict__ mods,
                            float* __restrict__ x1) {
    long t = (long)blockIdx.x * blockDim.x + threadIdx.x;
    if (t >= (long)MROWS * Cn) return;
    int c = (int)(t % Cn);
    int b = (int)(t / ((long)Gn * Cn));
    float v = raw[t] + mlp_b[c];
    x1[t] = x[t] + mods[b * 6 * Cn + 2 * Cn + c] * silu_f(v);
}

// ---------------- depthwise causal conv (dir selects sequence order) + silu ----------------
__global__ void conv_kernel(const float* __restrict__ xz, const float* __restrict__ cw,
                            const float* __restrict__ cb, float* __restrict__ uc, int dir) {
    long t = (long)blockIdx.x * blockDim.x + threadIdx.x;  // over B*L*DIN (scan order)
    if (t >= (long)Bn * Gn * DINn) return;
    int d = (int)(t % DINn);
    int l = (int)((t / DINn) % Gn);
    int b = (int)(t / ((long)DINn * Gn));
    float acc = cb[d];
#pragma unroll
    for (int j = 0; j < DCONVn; ++j) {
        int ls = l - (DCONVn - 1) + j;
        if (ls < 0) continue;
        int lnat = dir ? (Gn - 1 - ls) : ls;
        acc += cw[d * DCONVn + j] * xz[((long)b * Gn + lnat) * (2 * DINn) + d]; // u = xz[..., :DIN]
    }
    uc[t] = silu_f(acc);
}

// ---------------- selective scan: per-channel 16-state recurrence over L ----------------
__global__ void scan_kernel(const float* __restrict__ uc, const float* __restrict__ xd,
                            const float* __restrict__ dtraw, const float* __restrict__ db,
                            const float* __restrict__ Alog, const float* __restrict__ Dv,
                            const float* __restrict__ xz, float* __restrict__ ysum, int dir) {
    int b = blockIdx.x / (DINn / 256);
    int d = (blockIdx.x % (DINn / 256)) * 256 + threadIdx.x;
    __shared__ float Bs[DSTATEn], Cs[DSTATEn];
    float A[DSTATEn], h[DSTATEn];
#pragma unroll
    for (int s = 0; s < DSTATEn; ++s) { A[s] = -expf(Alog[d * DSTATEn + s]); h[s] = 0.f; }
    float dvd = Dv[d], dbd = db[d];
    for (int l = 0; l < Gn; ++l) {
        __syncthreads();
        if (threadIdx.x < 2 * DSTATEn) {
            float v = xd[((long)b * Gn + l) * XDP + DTRANKn + threadIdx.x];
            if (threadIdx.x < DSTATEn) Bs[threadIdx.x] = v; else Cs[threadIdx.x - DSTATEn] = v;
        }
        __syncthreads();
        float dtv = dtraw[((long)b * Gn + l) * DINn + d] + dbd;
        dtv = (dtv > 20.f) ? dtv : log1pf(expf(dtv));     // softplus
        float u = uc[((long)b * Gn + l) * DINn + d];
        float du = dtv * u;
        float y = 0.f;
#pragma unroll
        for (int s = 0; s < DSTATEn; ++s) {
            h[s] = expf(dtv * A[s]) * h[s] + du * Bs[s];
            y += h[s] * Cs[s];
        }
        int lnat = dir ? (Gn - 1 - l) : l;
        float z = xz[((long)b * Gn + lnat) * (2 * DINn) + DINn + d];
        float out = (y + dvd * u) * silu_f(z);
        long o = ((long)b * Gn + lnat) * DINn + d;
        if (dir) ysum[o] += out; else ysum[o] = out;
    }
}

// ---------------- epilogue 2: out = x1 + g_mlp * yo ----------------
__global__ void epi2_kernel(const float* __restrict__ x1, const float* __restrict__ yo,
                            const float* __restrict__ mods, float* __restrict__ out) {
    long t = (long)blockIdx.x * blockDim.x + threadIdx.x;
    if (t >= (long)MROWS * Cn) return;
    int c = (int)(t % Cn);
    int b = (int)(t / ((long)Gn * Cn));
    out[t] = x1[t] + mods[b * 6 * Cn + 5 * Cn + c] * yo[t];
}

// ================= host launcher =================
extern "C" void kernel_launch(void* const* d_in, const int* in_sizes, int n_in,
                              void* d_out, int out_size, void* d_ws, size_t ws_size,
                              hipStream_t stream) {
    const float* center    = (const float*)d_in[0];
    const float* x         = (const float*)d_in[1];
    const float* temd      = (const float*)d_in[2];
    const float* norm1_g   = (const float*)d_in[3];
    const float* norm1_b   = (const float*)d_in[4];
    const float* norm2_g   = (const float*)d_in[5];
    const float* norm2_b   = (const float*)d_in[6];
    const float* adaln_w   = (const float*)d_in[7];
    const float* adaln_b   = (const float*)d_in[8];
    const float* alpha     = (const float*)d_in[9];
    const float* beta      = (const float*)d_in[10];
    const float* prenorm_g = (const float*)d_in[11];
    const float* prenorm_b = (const float*)d_in[12];
    const float* mlp_w     = (const float*)d_in[13];
    const float* mlp_b     = (const float*)d_in[14];
    const float* in_proj_w = (const float*)d_in[15];
    const float* conv_w    = (const float*)d_in[16];
    const float* conv_b    = (const float*)d_in[17];
    const float* xproj_w   = (const float*)d_in[18];
    const float* dtproj_w  = (const float*)d_in[19];
    const float* dtproj_b  = (const float*)d_in[20];
    const float* A_log     = (const float*)d_in[21];
    const float* Dvec      = (const float*)d_in[22];
    const float* conv_wb   = (const float*)d_in[23];
    const float* conv_bb   = (const float*)d_in[24];
    const float* xproj_wb  = (const float*)d_in[25];
    const float* dtproj_wb = (const float*)d_in[26];
    const float* dtproj_bb = (const float*)d_in[27];
    const float* A_logb    = (const float*)d_in[28];
    const float* Dvecb     = (const float*)d_in[29];
    const float* out_proj_w= (const float*)d_in[30];
    float* out = (float*)d_out;

    // workspace carve-up (floats; all sizes multiple of 4 -> 16B alignment kept)
    float* W = (float*)d_ws;
    size_t off = 0;
    auto alloc = [&](size_t n) { float* p = W + off; off += n; return p; };
    float* mods    = alloc((size_t)Bn * 6 * Cn);
    float* h1      = alloc((size_t)MROWS * Cn);      // reused as h2
    float* lcn     = alloc((size_t)MROWS * D2n);
    float* x1      = alloc((size_t)MROWS * Cn);
    float* xz      = alloc((size_t)MROWS * 2 * DINn);
    float* uc      = alloc((size_t)MROWS * DINn);    // reused fwd/bwd
    float* xd      = alloc((size_t)MROWS * XDP);     // padded ld = 64
    float* dtraw   = alloc((size_t)MROWS * DINn);
    float* ysum    = alloc((size_t)MROWS * DINn);
    float* gemA    = alloc((size_t)MROWS * Cn);      // lfa raw, reused as yo
    _Float16* a16  = (_Float16*)alloc((size_t)MROWS * DINn / 2);   // f16 A staging (max 8192x768)
    _Float16* w16  = (_Float16*)alloc((size_t)Cn * 2 * DINn / 2);  // f16 W staging (max 384x1536)
    _Float16* xw16 = (_Float16*)alloc((size_t)DINn * XDP / 2);     // padded xproj  768x64
    _Float16* dw16 = (_Float16*)alloc((size_t)64 * DINn / 2);      // padded dtproj 64x768
    float* stats   = alloc(16);
    float* partial = alloc(2 * STAT_BLOCKS);
    int*   idx     = (int*)alloc((size_t)MROWS * Kn);
    (void)ws_size; (void)in_sizes; (void)n_in; (void)out_size;

    auto cvt = [&](const float* src, _Float16* dst, long n) {
        cvt_f16_kernel<<<(unsigned)((n / 4 + 255) / 256), 256, 0, stream>>>(src, dst, n);
    };

    // 1) adaLN mods
    mod_kernel<<<(Bn * 6 * Cn + 255) / 256, 256, 0, stream>>>(temd, adaln_w, adaln_b, mods);
    // 2) h1 = LN(x)*(1+sc_msa)+sh_msa  (chunks: sh=0, sc=1)
    lnmod_kernel<<<MROWS / 8, 256, 0, stream>>>(x, norm1_g, norm1_b, mods, 0, 1, h1);
    // 3) KNN
    knn_kernel<<<dim3(Gn / 256, Bn), 256, 0, stream>>>(center, idx);
    // 4) global std of diff
    diffstat_kernel<<<STAT_BLOCKS, 256, 0, stream>>>(h1, idx, partial);
    diffstat_final<<<1, 256, 0, stream>>>(partial, stats);
    // 5) local aggregation + prenorm LN
    lc_kernel<<<MROWS, 256, 0, stream>>>(h1, idx, stats, alpha, beta, prenorm_g, prenorm_b, lcn);
    // 6) lfa raw = lcn @ mlp_w  [8192x768]*[768x384]
    cvt(lcn, a16, (long)MROWS * D2n);
    cvt(mlp_w, w16, (long)D2n * Cn);
    gemm_wmma_kernel<<<dim3(Cn / 16, MROWS / BM), 256, 0, stream>>>(a16, D2n, w16, gemA, Cn, Cn, D2n);
    // 7) x1 = x + g_msa * silu(raw + mlp_b)
    epi1_kernel<<<(MROWS * Cn + 255) / 256, 256, 0, stream>>>(x, gemA, mlp_b, mods, x1);
    // 8) h2 = LN(x1)*(1+sc_mlp)+sh_mlp  (chunks: sh=3, sc=4), alias h1 buffer
    lnmod_kernel<<<MROWS / 8, 256, 0, stream>>>(x1, norm2_g, norm2_b, mods, 3, 4, h1);
    // 9) xz = h2 @ in_proj_w  [8192x384]*[384x1536]
    cvt(h1, a16, (long)MROWS * Cn);
    cvt(in_proj_w, w16, (long)Cn * 2 * DINn);
    gemm_wmma_kernel<<<dim3(2 * DINn / 16, MROWS / BM), 256, 0, stream>>>(a16, Cn, w16, xz, 2 * DINn, 2 * DINn, Cn);
    // 10) bidirectional SSM
    for (int dir = 0; dir < 2; ++dir) {
        const float* cw = dir ? conv_wb   : conv_w;
        const float* cb = dir ? conv_bb   : conv_b;
        const float* xw = dir ? xproj_wb  : xproj_w;
        const float* dw = dir ? dtproj_wb : dtproj_w;
        const float* db = dir ? dtproj_bb : dtproj_b;
        const float* Al = dir ? A_logb    : A_log;
        const float* Dv = dir ? Dvecb     : Dvec;
        pad_xw_kernel<<<(DINn * XDP + 255) / 256, 256, 0, stream>>>(xw, xw16);
        pad_dw_kernel<<<(64 * DINn + 255) / 256, 256, 0, stream>>>(dw, dw16);
        conv_kernel<<<(MROWS * DINn + 255) / 256, 256, 0, stream>>>(xz, cw, cb, uc, dir);
        // xd = uc @ xw16  [8192x768]*[768x64]
        cvt(uc, a16, (long)MROWS * DINn);
        gemm_wmma_kernel<<<dim3(XDP / 16, MROWS / BM), 256, 0, stream>>>(a16, DINn, xw16, xd, XDP, XDP, DINn);
        // dtraw = xd(:, 0:64) @ dw16  [8192x64]*[64x768]  (zero W rows 24..63 mask Bm/Cm cols)
        cvt(xd, a16, (long)MROWS * XDP);
        gemm_wmma_kernel<<<dim3(DINn / 16, MROWS / BM), 256, 0, stream>>>(a16, XDP, dw16, dtraw, DINn, DINn, KS);
        scan_kernel<<<Bn * (DINn / 256), 256, 0, stream>>>(uc, xd, dtraw, db, Al, Dv, xz, ysum, dir);
    }
    // 11) yo = ysum @ out_proj_w  [8192x768]*[768x384]
    cvt(ysum, a16, (long)MROWS * DINn);
    cvt(out_proj_w, w16, (long)DINn * Cn);
    gemm_wmma_kernel<<<dim3(Cn / 16, MROWS / BM), 256, 0, stream>>>(a16, DINn, w16, gemA, Cn, Cn, DINn);
    // 12) out = x1 + g_mlp * yo
    epi2_kernel<<<(MROWS * Cn + 255) / 256, 256, 0, stream>>>(x1, gemA, mods, out);
}
